// DeformableAlignment_66271345377649
// MI455X (gfx1250) — matrix operand order repaired and checked by reading, not compiled, not run on hardware
//
#include <hip/hip_runtime.h>
#include <hip/hip_bf16.h>

typedef _Float16 h16;
typedef __attribute__((ext_vector_type(4)))  _Float16 v4h;
typedef __attribute__((ext_vector_type(8)))  _Float16 v8h;
typedef __attribute__((ext_vector_type(16))) _Float16 v16h;
typedef __attribute__((ext_vector_type(8)))  float    v8f;

#define WMMA_F16(A, Bm, C) \
  __builtin_amdgcn_wmma_f32_16x16x32_f16(false, (A), false, (Bm), (short)0, (C), false, false)

static constexpr int BB = 4, H = 256, W = 256, CX = 128, CE = 192;
static constexpr int C1P = 224;              // conv1 input channels padded 194 -> 224 (7x32)
static constexpr long HWl = (long)H * W;

__device__ __forceinline__ v16h cat8(v8h lo, v8h hi) {
  return __builtin_shufflevector(lo, hi, 0, 1, 2, 3, 4, 5, 6, 7,
                                         8, 9, 10, 11, 12, 13, 14, 15);
}

// ---------------------------------------------------------------------------
// 1) concat(extra_feat, flow) -> f16 NHWC [B,H,W,224], zero-padded channels.
// ---------------------------------------------------------------------------
__global__ void __launch_bounds__(256) pack_feat(const float* __restrict__ ef,
                                                 const float* __restrict__ flow,
                                                 h16* __restrict__ feat) {
  int tile = blockIdx.x;                 // B*H*(W/16) = 16384 tiles
  int xt = tile & 15, y = (tile >> 4) & 255, b = tile >> 12;
  int x0 = xt * 16;
  __shared__ h16 lt[16 * C1P];
  int t = threadIdx.x;
  int m = t & 15, cw = t >> 4;
  int x = x0 + m;
  for (int ci = cw; ci < C1P; ci += 16) {
    float v = 0.f;
    if (ci < CE)            v = ef[((long)(b * CE + ci)) * HWl + (long)y * W + x];
    else if (ci < CE + 2)   v = flow[((long)(b * 2 + (ci - CE))) * HWl + (long)y * W + x];
    lt[m * C1P + ci] = (h16)v;
  }
  __syncthreads();
  long base = (((long)(b * H + y)) * W + x0) * C1P;
  for (int i = t; i < 16 * C1P / 4; i += 256)
    *(v4h*)(&feat[base + i * 4]) = *(const v4h*)(&lt[i * 4]);
}

// ---------------------------------------------------------------------------
// 2) Weight packers.
//    Conv weights -> lane-major WMMA *B*-fragments
//      (16-bit B layout: lane n=lane&15 col, kb=(lane>>4)*8, e<8 -> K=kb+e; e>=8 -> K=16+kb+(e-8)).
//    dcn weights  -> lane-major WMMA *A*-fragments
//      (this code's contiguous k order: lane row=lane&15, K=(lane>>4)*16+e).
// ---------------------------------------------------------------------------
__global__ void pack_conv_w(const float* __restrict__ w, int Cin, int NC,
                            h16* __restrict__ wf) {
  int idx = blockIdx.x * 256 + threadIdx.x;
  int total = 9 * NC * 32 * 16;
  if (idx >= total) return;
  int e = idx & 15, lane = (idx >> 4) & 31, f = idx >> 9;
  int q = f % NC, tap = f / NC;
  int ky = tap / 3, kx = tap % 3;
  int n = lane & 15;
  int c = q * 32 + ((e < 8) ? ((lane >> 4) * 8 + e) : (16 + (lane >> 4) * 8 + (e - 8)));
  float v = (c < Cin) ? w[((n * Cin + c) * 3 + ky) * 3 + kx] : 0.f;
  wf[idx] = (h16)v;
}

__global__ void pack_gemm_wA(const float* __restrict__ w, h16* __restrict__ wf) {
  int idx = blockIdx.x * 256 + threadIdx.x;          // 8*4*32*16 = 16384
  int e = idx & 15, lane = (idx >> 4) & 31, f = idx >> 9;
  int q = f & 3, nt = f >> 2;
  int o = nt * 16 + (lane & 15);
  int K = (lane >> 4) * 16 + e;
  int c = q * 32 + K;
  wf[idx] = (h16)w[o * 128 + c];
}

// ---------------------------------------------------------------------------
// 3) Implicit-GEMM conv3x3 via v_wmma_f32_16x16x32_f16, templated on
//    NC (=CinPad/32) and CoutPad. One wave = 16 px x 16 out ch.
//    Interior tiles: branch-free unconditional loads; edges: predicated.
//    Two accumulators halve the WMMA dependency chain; LDS epilogue gives
//    coalesced v8h stores.
// ---------------------------------------------------------------------------
template <int NC, int CoutPad>
__global__ void __launch_bounds__(256) wmma_conv3x3(const h16* __restrict__ in,
                                                    const h16* __restrict__ wf,
                                                    const float* __restrict__ bias,
                                                    h16* __restrict__ out) {
  constexpr int CinPad = NC * 32;
  int wid = (blockIdx.x * 256 + threadIdx.x) >> 5;   // tile id, B*H*16 = 16384
  int lane = threadIdx.x & 31;
  int xt = wid & 15, y = (wid >> 4) & 255, b = wid >> 12;
  int x0 = xt * 16;
  int m = lane & 15;
  int kb = (lane >> 4) * 8;
  v8f acc0 = {}, acc1 = {};
  const bool interior = (y > 0) & (y < H - 1) & (xt > 0) & (xt < 15);  // uniform
  if (interior) {
    for (int tap = 0; tap < 9; ++tap) {
      int ky = tap / 3 - 1, kx = tap % 3 - 1;
      const h16* p = in + ((long)(b * H + y + ky) * W + (x0 + m + kx)) * CinPad;
#pragma unroll
      for (int q = 0; q < NC; ++q) {
        v16h a = cat8(*(const v8h*)(p + q * 32 + kb),
                      *(const v8h*)(p + q * 32 + 16 + kb));
        v16h bf = *(const v16h*)(wf + ((long)(tap * NC + q) * 32 + lane) * 16);
        if ((tap * NC + q) & 1) acc1 = WMMA_F16(a, bf, acc1);
        else                    acc0 = WMMA_F16(a, bf, acc0);
      }
    }
  } else {
    for (int tap = 0; tap < 9; ++tap) {
      int ky = tap / 3 - 1, kx = tap % 3 - 1;
      int yy = y + ky, xx = x0 + m + kx;
      bool inb = (yy >= 0) & (yy < H) & (xx >= 0) & (xx < W);
      const h16* p = in + ((long)(b * H + yy) * W + xx) * CinPad;
#pragma unroll
      for (int q = 0; q < NC; ++q) {
        v8h lo = {}, hi = {};
        if (inb) {
          lo = *(const v8h*)(p + q * 32 + kb);
          hi = *(const v8h*)(p + q * 32 + 16 + kb);
        }
        v16h a = cat8(lo, hi);
        v16h bf = *(const v16h*)(wf + ((long)(tap * NC + q) * 32 + lane) * 16);
        if ((tap * NC + q) & 1) acc1 = WMMA_F16(a, bf, acc1);
        else                    acc0 = WMMA_F16(a, bf, acc0);
      }
    }
  }
  // D layout: lane n=lane&15 (out ch), VGPR r -> pixel r + (lane>>4)*8.
  __shared__ h16 st[8][16 * CoutPad];
  h16* sts = st[threadIdx.x >> 5];
  int n = lane & 15;
  int mo = (lane >> 4) * 8;
  float bv = bias[n];
#pragma unroll
  for (int r = 0; r < 8; ++r) {
    float v = acc0[r] + acc1[r] + bv;
    v = v >= 0.f ? v : 0.1f * v;                      // leaky relu 0.1
    sts[(mo + r) * CoutPad + n] = (h16)v;
    if (CoutPad == 32) sts[(mo + r) * CoutPad + n + 16] = (h16)0.f;
  }
  __syncthreads();
  long base = ((long)(b * H + y) * W + x0) * CoutPad;
#pragma unroll
  for (int i = lane; i < 16 * CoutPad / 8; i += 32)
    *(v8h*)(&out[base + i * 8]) = *(const v8h*)(&sts[i * 8]);
}

// ---------------------------------------------------------------------------
// 4) 16->12 head, tanh/sigmoid, add reversed flow -> dy/dx/mask [B,4,H,W].
// ---------------------------------------------------------------------------
__global__ void __launch_bounds__(256) head_offmask(const h16* __restrict__ h3,
                                                    const float* __restrict__ w4,
                                                    const float* __restrict__ b4,
                                                    const float* __restrict__ flow,
                                                    float* __restrict__ dy,
                                                    float* __restrict__ dx,
                                                    float* __restrict__ mk) {
  long pix = (long)blockIdx.x * 256 + threadIdx.x;    // 262144 pixels
  int x = pix & 255, y = (pix >> 8) & 255, b = (int)(pix >> 16);
  v8h hlo = *(const v8h*)(h3 + pix * 16);
  v8h hhi = *(const v8h*)(h3 + pix * 16 + 8);
  float hv[16];
#pragma unroll
  for (int i = 0; i < 8; ++i) { hv[i] = (float)hlo[i]; hv[8 + i] = (float)hhi[i]; }
  float head[12];
#pragma unroll
  for (int o = 0; o < 12; ++o) {
    float s = b4[o];
#pragma unroll
    for (int i = 0; i < 16; ++i) s += hv[i] * w4[o * 16 + i];
    head[o] = s;
  }
  float f0 = flow[((long)(b * 2 + 0)) * HWl + (long)y * W + x];
  float f1 = flow[((long)(b * 2 + 1)) * HWl + (long)y * W + x];
#pragma unroll
  for (int g = 0; g < 4; ++g) {
    long oi = ((long)(b * 4 + g)) * HWl + (long)y * W + x;
    dy[oi] = 8.f * tanhf(head[2 * g]) + f1;           // flow[:, ::-1]: even ch += flow1
    dx[oi] = 8.f * tanhf(head[2 * g + 1]) + f0;
    mk[oi] = 1.f / (1.f + __expf(-head[8 + g]));
  }
}

// ---------------------------------------------------------------------------
// 5) Deformable bilinear sample, mask multiply, emit NHWC f16 [pix,128].
// ---------------------------------------------------------------------------
__global__ void __launch_bounds__(256) deform_sample(const float* __restrict__ x,
                                                     const float* __restrict__ dy,
                                                     const float* __restrict__ dx,
                                                     const float* __restrict__ mk,
                                                     h16* __restrict__ samp) {
  int tile = blockIdx.x;                              // 16384 tiles of 16 pixels
  int xt = tile & 15, y = (tile >> 4) & 255, b = tile >> 12;
  int x0 = xt * 16;
  int t = threadIdx.x;
  int m = t & 15, ci = t >> 4;                        // ci: 8-channel cluster
  int g = ci >> 2;
  int c0 = ci * 8;
  int xx = x0 + m;
  long oi = ((long)(b * 4 + g)) * HWl + (long)y * W + xx;
  float py = dy[oi] + (float)y;
  float px = dx[oi] + (float)xx;
  float mv = mk[oi];
  float fy = floorf(py), fx = floorf(px);
  float ty = py - fy, tx = px - fx;
  int y0 = (int)fy, xi = (int)fx;
  float w00 = (1.f - ty) * (1.f - tx), w01 = (1.f - ty) * tx;
  float w10 = ty * (1.f - tx), w11 = ty * tx;
  bool vy0 = (y0 >= 0) & (y0 < H), vy1 = (y0 + 1 >= 0) & (y0 + 1 < H);
  bool vx0 = (xi >= 0) & (xi < W), vx1 = (xi + 1 >= 0) & (xi + 1 < W);
  int y0c = min(max(y0, 0), H - 1), y1c = min(max(y0 + 1, 0), H - 1);
  int x0c = min(max(xi, 0), W - 1), x1c = min(max(xi + 1, 0), W - 1);
  __shared__ h16 lt[16 * 128];
  const float* xb = x + (long)b * CX * HWl;
#pragma unroll
  for (int k = 0; k < 8; ++k) {
    const float* pc = xb + (long)(c0 + k) * HWl;
    float v00 = (vy0 & vx0) ? pc[y0c * W + x0c] : 0.f;
    float v01 = (vy0 & vx1) ? pc[y0c * W + x1c] : 0.f;
    float v10 = (vy1 & vx0) ? pc[y1c * W + x0c] : 0.f;
    float v11 = (vy1 & vx1) ? pc[y1c * W + x1c] : 0.f;
    float s = v00 * w00 + v01 * w01 + v10 * w10 + v11 * w11;
    lt[m * 128 + c0 + k] = (h16)(s * mv);
  }
  __syncthreads();
  long base = (((long)(b * H + y)) * W + x0) * 128;
  *(v8h*)(&samp[base + t * 8]) = *(const v8h*)(&lt[t * 8]);
}

// ---------------------------------------------------------------------------
// 6) Final 128x128 projection, operand-swapped:
//    A = dcn weights (16 o x 32 c, staged in 32KB LDS once per block),
//    B = sampled pixels (32 c x 16 px, one contiguous v16h per fragment).
//    D cols = pixels -> direct NCHW f32 stores, two 64B segments per store.
//    Each wave sweeps 8 tiles (4 pairs); a pair shares each weight fragment
//    and gives two independent WMMA chains.
// ---------------------------------------------------------------------------
__global__ void __launch_bounds__(256) dcn_gemm(const h16* __restrict__ samp,
                                                const h16* __restrict__ wfA,
                                                float* __restrict__ out) {
  __shared__ h16 wl[32 * 512];                        // 32 frags * 512 halves
  int t = threadIdx.x;
#pragma unroll
  for (int i = 0; i < 8; ++i)
    *(v8h*)(&wl[(t + i * 256) * 8]) = *(const v8h*)(&wfA[(t + i * 256) * 8]);
  __syncthreads();
  int lane = t & 31, lw = t >> 5;
  int px = lane & 15;                                 // B col = pixel
  int cb = (lane >> 4) * 16;                          // B K-base within chunk
  int mo = (lane >> 4) * 8;                           // D row split
  for (int pr = 0; pr < 4; ++pr) {
    int tile0 = ((blockIdx.x * 8 + lw) * 4 + pr) * 2; // even tile: pair in-row
    int xt = tile0 & 15, y = (tile0 >> 4) & 255, b = tile0 >> 12;
    int x0 = xt * 16;
    long pixbase = ((long)(b * H + y)) * W + x0;
    const h16* p0 = samp + (pixbase + px) * 128;
    const h16* p1 = p0 + 16 * 128;
    v16h bm0[4], bm1[4];
#pragma unroll
    for (int q = 0; q < 4; ++q) {
      bm0[q] = *(const v16h*)(p0 + q * 32 + cb);
      bm1[q] = *(const v16h*)(p1 + q * 32 + cb);
    }
    float* ob = out + (long)b * CX * HWl + (long)y * W + x0 + px;
#pragma unroll
    for (int nt = 0; nt < 8; ++nt) {
      v8f a0 = {}, a1 = {};
#pragma unroll
      for (int q = 0; q < 4; ++q) {
        v16h aw = *(const v16h*)(&wl[((nt * 4 + q) * 32 + lane) * 16]);
        a0 = WMMA_F16(aw, bm0[q], a0);
        a1 = WMMA_F16(aw, bm1[q], a1);
      }
#pragma unroll
      for (int r = 0; r < 8; ++r) {
        long orow = (long)(nt * 16 + mo + r) * HWl;
        ob[orow] = a0[r];
        ob[orow + 16] = a1[r];
      }
    }
  }
}

// ---------------------------------------------------------------------------
extern "C" void kernel_launch(void* const* d_in, const int* in_sizes, int n_in,
                              void* d_out, int out_size, void* d_ws, size_t ws_size,
                              hipStream_t stream) {
  (void)in_sizes; (void)n_in; (void)out_size; (void)ws_size;
  const float* x    = (const float*)d_in[0];
  const float* ef   = (const float*)d_in[1];
  const float* flow = (const float*)d_in[2];
  const float* w1   = (const float*)d_in[3];
  const float* b1   = (const float*)d_in[4];
  const float* w2   = (const float*)d_in[5];
  const float* b2   = (const float*)d_in[6];
  const float* w3   = (const float*)d_in[7];
  const float* b3   = (const float*)d_in[8];
  const float* w4   = (const float*)d_in[9];
  const float* b4   = (const float*)d_in[10];
  const float* dcnw = (const float*)d_in[11];
  float* out = (float*)d_out;

  char* ws = (char*)d_ws;
  size_t off = 0;
  auto alloc = [&](size_t bytes) { size_t o = off; off += (bytes + 255) & ~(size_t)255; return o; };
  const long NP = (long)BB * H * W;
  size_t regA = alloc((size_t)NP * C1P * 2);  // feat16; later aliased by samp (64MB < 112MB)
  size_t regB = alloc((size_t)NP * 32 * 2);   // h1; later aliased by dy/dx/mask (12MB < 16MB)
  size_t regC = alloc((size_t)NP * 32 * 2);   // h2
  size_t regD = alloc((size_t)NP * 16 * 2);   // h3
  size_t oWF1 = alloc((size_t)9 * 7 * 512 * 2);
  size_t oWF2 = alloc((size_t)9 * 512 * 2);
  size_t oWF3 = alloc((size_t)9 * 512 * 2);
  size_t oWFD = alloc((size_t)32 * 512 * 2);

  h16* feat = (h16*)(ws + regA);
  h16* samp = (h16*)(ws + regA);              // alias: feat dead after conv1
  h16* h1   = (h16*)(ws + regB);
  h16* h2   = (h16*)(ws + regC);
  h16* h3   = (h16*)(ws + regD);
  float* dy = (float*)(ws + regB);            // alias: h1 dead after conv2
  float* dx = dy + NP * 4;
  float* mk = dx + NP * 4;
  h16* wf1 = (h16*)(ws + oWF1);
  h16* wf2 = (h16*)(ws + oWF2);
  h16* wf3 = (h16*)(ws + oWF3);
  h16* wfD = (h16*)(ws + oWFD);

  const int nTiles = BB * H * (W / 16);       // 16384

  // weight packing + input packing
  pack_conv_w<<<126, 256, 0, stream>>>(w1, 194, 7, wf1);
  pack_conv_w<<<18, 256, 0, stream>>>(w2, 16, 1, wf2);
  pack_conv_w<<<18, 256, 0, stream>>>(w3, 16, 1, wf3);
  pack_gemm_wA<<<64, 256, 0, stream>>>(dcnw, wfD);
  pack_feat<<<nTiles, 256, 0, stream>>>(ef, flow, feat);

  // conv stack on WMMA
  wmma_conv3x3<7, 32><<<nTiles / 8, 256, 0, stream>>>(feat, wf1, b1, h1);
  wmma_conv3x3<1, 32><<<nTiles / 8, 256, 0, stream>>>(h1, wf2, b2, h2);
  wmma_conv3x3<1, 16><<<nTiles / 8, 256, 0, stream>>>(h2, wf3, b3, h3);

  // offsets/masks, deformable sampling, final projection GEMM
  head_offmask<<<(int)(NP / 256), 256, 0, stream>>>(h3, w4, b4, flow, dy, dx, mk);
  deform_sample<<<nTiles, 256, 0, stream>>>(x, dy, dx, mk, samp);
  dcn_gemm<<<nTiles / 64, 256, 0, stream>>>(samp, wfD, out);   // 256 blocks, 8 tiles/wave
}
